// SilhouetteLoss_70257075028594
// MI455X (gfx1250) — matrix operand order repaired and checked by reading, not compiled, not run on hardware
//
#include <hip/hip_runtime.h>

typedef __attribute__((ext_vector_type(16))) __bf16 v16bf;
typedef __attribute__((ext_vector_type(8)))  float  v8f;

#define N_PTS   8192
#define D_FEATD 256
#define N_CLU   64
#define ROWS_PER_BLK 32
#define BIN_PITCH 65              // 64 clusters + 1 pad: kills cross-half bank conflicts
#define COL_TILES (N_PTS / 16)

union FragBF { uint4 u[2]; v16bf v; };

struct __attribute__((aligned(8))) SqLab { float sq; int lab; };

// ------ Kernel 1: f32 -> bf16 copy + squared norms + {sq,label} pack ---------
__global__ void prep_kernel(const float* __restrict__ x,
                            const int* __restrict__ labels,
                            unsigned short* __restrict__ xbf,
                            SqLab* __restrict__ sqlab) {
  int wave = (int)((blockIdx.x * blockDim.x + threadIdx.x) >> 5);
  int lane = threadIdx.x & 31;
  if (wave >= N_PTS) return;
  const float* row = x + (size_t)wave * D_FEATD;
  unsigned short* orow = xbf + (size_t)wave * D_FEATD;
  float acc = 0.f;
  for (int k = lane; k < D_FEATD; k += 32) {
    float f = row[k];
    acc += f * f;
    union { __bf16 h; unsigned short s; } cv;
    cv.h = (__bf16)f;
    orow[k] = cv.s;
  }
  for (int off = 16; off > 0; off >>= 1)
    acc += __shfl_down(acc, off, 32);
  if (lane == 0) {
    SqLab sl;
    sl.sq = acc;
    sl.lab = labels[wave];
    sqlab[wave] = sl;
  }
}

// ------ Kernel 2: fused WMMA pairwise-dist + branchless per-cluster binning ---
__global__ void __launch_bounds__(256, 1)
silhouette_main(const unsigned short* __restrict__ xbf,
                const SqLab* __restrict__ sqlab,
                float* __restrict__ svals) {
  __shared__ float    s_sum[ROWS_PER_BLK * BIN_PITCH];   // 8.1 KB
  __shared__ unsigned s_min[ROWS_PER_BLK * BIN_PITCH];   // 8.1 KB (f32 bits)
  __shared__ unsigned s_csize[N_CLU];                    // cluster sizes

  const int tid = threadIdx.x;
  for (int i = tid; i < ROWS_PER_BLK * BIN_PITCH; i += 256) {
    s_sum[i] = 0.f;
    s_min[i] = 0x7f800000u;   // +inf
  }
  if (tid < N_CLU) s_csize[tid] = 0u;
  __syncthreads();

  // Cluster-size histogram (consumed only after the final barrier).
  for (int i = tid; i < N_PTS; i += 256)
    atomicAdd(&s_csize[sqlab[i].lab], 1u);

  const int wave = tid >> 5;
  const int lane = tid & 31;
  const int rowtile = wave >> 2;           // 2 row tiles of 16, 4 waves each
  const int phase   = wave & 3;            // 4-way column interleave
  const int row_base = blockIdx.x * ROWS_PER_BLK;
  const int sel = lane >> 4;               // lane half
  const int nl  = lane & 15;

  // --- Preload A fragments for this wave's 16 rows, full K=256 (8 chunks) ---
  // 16-bit A layout: lane m (0-15) K in {0..7,16..23}; lane m+16 K in {8..15,24..31}
  FragBF afrag[8];
  {
    const int grow = row_base + rowtile * 16 + nl;
    const unsigned short* rp = xbf + (size_t)grow * D_FEATD;
#pragma unroll
    for (int kc = 0; kc < 8; ++kc) {
      const unsigned short* p = rp + kc * 32 + sel * 8;   // halves
      afrag[kc].u[0] = *(const uint4*)(p);
      afrag[kc].u[1] = *(const uint4*)(p + 16);
    }
  }
  // sq of the 8 M-rows this lane-half contributes to in C/D layout
  float sqm[8];
#pragma unroll
  for (int v = 0; v < 8; ++v)
    sqm[v] = sqlab[row_base + rowtile * 16 + sel * 8 + v].sq;

  // --- Sweep all column tiles (4-way interleave among the wave quad) ---
  for (int ct = phase; ct < COL_TILES; ct += 4) {
    const int col_base = ct * 16;
    const int gcol = col_base + nl;
    const unsigned short* cp = xbf + (size_t)gcol * D_FEATD;

    // Unconditional near-cache prefetch of next tile (overrun lands in our
    // own workspace; speculative prefetch drops invalid translations).
    __builtin_prefetch(cp + 64 * D_FEATD, 0, 3);

    const SqLab sl = sqlab[gcol];          // single b64: {sq, label}

    // Two independent WMMA accumulation chains (even/odd K-chunks) so the
    // XDL pipe can overlap them instead of one 8-deep serial chain.
    v8f c0 = {}, c1 = {};
#pragma unroll
    for (int kc = 0; kc < 8; kc += 2) {
      // 16-bit B layout: lane n (0-15) K in {0..7,16..23}; lane n+16 K in {8..15,24..31}
      FragBF b0, b1;
      const unsigned short* bp0 = cp + kc * 32 + sel * 8;         // halves
      const unsigned short* bp1 = cp + (kc + 1) * 32 + sel * 8;
      b0.u[0] = *(const uint4*)(bp0);
      b0.u[1] = *(const uint4*)(bp0 + 16);
      b1.u[0] = *(const uint4*)(bp1);
      b1.u[1] = *(const uint4*)(bp1 + 16);
      c0 = __builtin_amdgcn_wmma_f32_16x16x32_bf16(
               false, afrag[kc].v,     false, b0.v, (short)0, c0, false, false);
      c1 = __builtin_amdgcn_wmma_f32_16x16x32_bf16(
               false, afrag[kc + 1].v, false, b1.v, (short)0, c1, false, false);
    }
    v8f c = c0 + c1;

    // d = sqrt(max(|xi|^2 + |xj|^2 - 2 xi.xj, 0)); diagonal forced to 0.
    // Fully branchless: sum-add of 0 is a no-op, min of 0 only pollutes the
    // own-cluster bin (ignored by the epilogue); counts are reconstructed
    // from cluster sizes (self-distance is the only zero).
#pragma unroll
    for (int v = 0; v < 8; ++v) {
      const int m_local = rowtile * 16 + sel * 8 + v;
      const int grow = row_base + m_local;
      float d2 = fmaxf(sqm[v] + sl.sq - 2.0f * c[v], 0.f);
      float d  = __builtin_amdgcn_sqrtf(d2);     // bare v_sqrt_f32
      d = (grow == gcol) ? 0.f : d;
      const int bin = m_local * BIN_PITCH + sl.lab;
      atomicMin(&s_min[bin], __float_as_uint(d));   // ds_min_u32
      atomicAdd(&s_sum[bin], d);                    // ds_add_f32
    }
  }
  __syncthreads();

  // --- Epilogue: finalize silhouette score per row owned by this block ---
  if (tid < ROWS_PER_BLK) {
    const int grow = row_base + tid;
    const int lab = sqlab[grow].lab;
    const float*    rs = &s_sum[tid * BIN_PITCH];
    const unsigned* rm = &s_min[tid * BIN_PITCH];

    float a = rs[lab] / (float)(s_csize[lab] - 1u);
    unsigned best = 0xffffffffu;
    int bc = 0;
    for (int cidx = 0; cidx < N_CLU; ++cidx) {
      if (cidx == lab) continue;
      unsigned mv = rm[cidx];
      if (mv < best) { best = mv; bc = cidx; }
    }
    float b = rs[bc] / (float)s_csize[bc];
    svals[grow] = (b - a) / fmaxf(a, b);
  }
}

// ------ Kernel 3: deterministic reduction -> -mean(s) -------------------------
__global__ void reduce_kernel(const float* __restrict__ svals,
                              float* __restrict__ out) {
  __shared__ float red[256];
  float acc = 0.f;
  for (int i = threadIdx.x; i < N_PTS; i += 256) acc += svals[i];
  red[threadIdx.x] = acc;
  __syncthreads();
  for (int off = 128; off > 0; off >>= 1) {
    if ((int)threadIdx.x < off) red[threadIdx.x] += red[threadIdx.x + off];
    __syncthreads();
  }
  if (threadIdx.x == 0) out[0] = -red[0] / (float)N_PTS;
}

extern "C" void kernel_launch(void* const* d_in, const int* in_sizes, int n_in,
                              void* d_out, int out_size, void* d_ws, size_t ws_size,
                              hipStream_t stream) {
  const float* feats  = (const float*)d_in[0];
  const int*   labels = (const int*)d_in[1];

  char* ws = (char*)d_ws;
  unsigned short* xbf = (unsigned short*)ws;                          // 4 MB
  SqLab* sqlab = (SqLab*)(ws + (size_t)N_PTS * D_FEATD * 2);          // 64 KB
  float* svals = (float*)(ws + (size_t)N_PTS * D_FEATD * 2 + N_PTS * 8);

  prep_kernel<<<N_PTS / 8, 256, 0, stream>>>(feats, labels, xbf, sqlab);
  silhouette_main<<<N_PTS / ROWS_PER_BLK, 256, 0, stream>>>(xbf, sqlab, svals);
  reduce_kernel<<<1, 256, 0, stream>>>(svals, (float*)d_out);
}
